// LocalAttention_14542759265043
// MI455X (gfx1250) — matrix operand order repaired and checked by reading, not compiled, not run on hardware
//
#include <hip/hip_runtime.h>
#include <hip/hip_bf16.h>
#include <math.h>

typedef __bf16 bf16;
typedef __attribute__((ext_vector_type(16))) bf16 v16bf;
typedef __attribute__((ext_vector_type(8)))  bf16 v8bf;
typedef __attribute__((ext_vector_type(8)))  float v8f;
typedef __attribute__((ext_vector_type(4)))  float v4f;

constexpr int kB = 8, kE = 256, kH = 80, kW = 80, kN = 6400, kNH = 8, kHD = 32;
constexpr int kNPIX = kB * kN;             // 51200 pixels total
constexpr float kBnEps = 1e-5f;

// ---------------------------------------------------------------- helpers
__device__ __forceinline__ v8f wmma_bf16(v16bf a, v16bf b, v8f c) {
  // D = A(16x32 bf16) * B(32x16 bf16) + C(16x16 f32)
  return __builtin_amdgcn_wmma_f32_16x16x32_bf16(false, a, false, b,
                                                 (short)0, c, false, false);
}

__device__ __forceinline__ v16bf join8(v8bf lo, v8bf hi) {
  v16bf r;
#pragma unroll
  for (int j = 0; j < 8; ++j) { r[j] = lo[j]; r[j + 8] = hi[j]; }
  return r;
}

__device__ __forceinline__ float gelu_exact(float x) {
  return 0.5f * x * (1.0f + erff(x * 0.70710678118654752440f));
}

// LDS byte offset of a generic pointer into a __shared__ array
typedef __attribute__((address_space(3))) unsigned char lds_u8;
__device__ __forceinline__ unsigned int lds_off_of(void* p) {
  return (unsigned int)(unsigned long long)(lds_u8*)p;
}

// CDNA5 async global->LDS copies (ASYNCcnt path, no VGPR data staging).
// Same INST_OFFSET applies to both LDS and global sides (ISA 10, async ops).
__device__ __forceinline__ void async_g2l_32B(unsigned int lds_off,
                                              const void* gptr) {
  unsigned long long ga = (unsigned long long)gptr;
  asm volatile(
      "global_load_async_to_lds_b128 %0, %1, off\n\t"
      "global_load_async_to_lds_b128 %0, %1, off offset:16"
      :: "v"(lds_off), "v"(ga) : "memory");
}
__device__ __forceinline__ void async_g2l_64B(unsigned int lds_off,
                                              const void* gptr) {
  unsigned long long ga = (unsigned long long)gptr;
  asm volatile(
      "global_load_async_to_lds_b128 %0, %1, off\n\t"
      "global_load_async_to_lds_b128 %0, %1, off offset:16\n\t"
      "global_load_async_to_lds_b128 %0, %1, off offset:32\n\t"
      "global_load_async_to_lds_b128 %0, %1, off offset:48"
      :: "v"(lds_off), "v"(ga) : "memory");
}
__device__ __forceinline__ void wait_async_all() {
  asm volatile("s_wait_asynccnt 0x0" ::: "memory");
}

// ------------------------------------------------------- small prep kernels
__global__ __launch_bounds__(256) void k_cvt_weights(
    const float* __restrict__ pw, const float* __restrict__ qkv,
    bf16* __restrict__ wpw, bf16* __restrict__ wqkv) {
  const int npw = 3 * kE * kE;               // 196608
  const int nqk = 3 * 3 * kE * (kE / kNH);   // 73728
  int i = blockIdx.x * blockDim.x + threadIdx.x;
  if (i < npw)            wpw[i] = (bf16)pw[i];
  else if (i < npw + nqk) wqkv[i - npw] = (bf16)qkv[i - npw];
}

__global__ __launch_bounds__(256) void k_pool(const float* __restrict__ x,
                                              float* __restrict__ pooled) {
  __shared__ float s[256];
  const int be = blockIdx.x;                 // b*E + e
  const float* p = x + (size_t)be * kN;
  float acc = 0.f;
  for (int i = threadIdx.x; i < kN; i += 256) acc += p[i];
  s[threadIdx.x] = acc;
  __syncthreads();
  for (int st = 128; st > 0; st >>= 1) {
    if (threadIdx.x < st) s[threadIdx.x] += s[threadIdx.x + st];
    __syncthreads();
  }
  if (threadIdx.x == 0) pooled[be] = s[0] * (1.0f / kN);
}

__global__ __launch_bounds__(256) void k_dyn(const float* __restrict__ pooled,
                                             const float* __restrict__ fcw,
                                             const float* __restrict__ fcb,
                                             float* __restrict__ dyn) {
  __shared__ float s[3][256];
  const int b = blockIdx.x;
  const int t = threadIdx.x;                 // == channel e
  const float pv = pooled[b * kE + t];
#pragma unroll
  for (int i = 0; i < 3; ++i) s[i][t] = fcw[i * kE + t] * pv;
  __syncthreads();
  for (int st = 128; st > 0; st >>= 1) {
    if (t < st) {
#pragma unroll
      for (int i = 0; i < 3; ++i) s[i][t] += s[i][t + st];
    }
    __syncthreads();
  }
  if (t == 0) {
    float l0 = s[0][0] + fcb[0], l1 = s[1][0] + fcb[1], l2 = s[2][0] + fcb[2];
    float m = fmaxf(l0, fmaxf(l1, l2));
    float e0 = expf(l0 - m), e1 = expf(l1 - m), e2 = expf(l2 - m);
    float inv = 1.0f / (e0 + e1 + e2);
    dyn[b * 3 + 0] = e0 * inv;
    dyn[b * 3 + 1] = e1 * inv;
    dyn[b * 3 + 2] = e2 * inv;
  }
}

__global__ __launch_bounds__(256) void k_init_out(const float* __restrict__ x,
                                                  float* __restrict__ out) {
  int i = blockIdx.x * blockDim.x + threadIdx.x;
  ((v4f*)out)[i] = ((const v4f*)x)[i];
}

// ------------------------------------------------ depthwise conv (per branch)
// writes h in (pixel-major, channel-contiguous) bf16 layout for the GEMMs
template <int KS>
__global__ __launch_bounds__(256) void k_dwconv(
    const float* __restrict__ x, const float* __restrict__ pos,
    const float* __restrict__ dww, const float* __restrict__ dwb,
    bf16* __restrict__ hbf) {
  const int idx = blockIdx.x * blockDim.x + threadIdx.x;  // (b,e,y,x)
  const int xc = idx % kW;
  const int y  = (idx / kW) % kH;
  const int e  = (idx / (kW * kH)) % kE;
  const int b  = idx / (kW * kH * kE);
  constexpr int P = KS / 2;
  const float pe = pos[e];
  const float* xp = x + (size_t)(b * kE + e) * kN;
  const float* wk = dww + e * KS * KS;
  float acc = dwb[e];
#pragma unroll
  for (int ky = 0; ky < KS; ++ky) {
    const int yy = y + ky - P;
    if (yy < 0 || yy >= kH) continue;
#pragma unroll
    for (int kx = 0; kx < KS; ++kx) {
      const int xx = xc + kx - P;
      if (xx < 0 || xx >= kW) continue;
      acc += (xp[yy * kW + xx] + pe) * wk[ky * KS + kx];
    }
  }
  hbf[(size_t)(b * kN + y * kW + xc) * kE + e] = (bf16)acc;
}

// ------------------------- pointwise 1x1 (E x E GEMM) + bias + BN + GELU
// block: 256 thr = 8 waves; tile: 128 couts (blockIdx.y half) x 32 pixels.
// Activation tile (32 px x 256 ch bf16 = 16KB) async-copied to LDS, shared
// by all 8 waves; each wave runs 16 WMMAs (2 pixel subtiles share A-frags).
__global__ __launch_bounds__(256) void k_pwgemm(
    const bf16* __restrict__ hbf, const bf16* __restrict__ wpw,
    const float* __restrict__ pwb,
    const float* __restrict__ bng, const float* __restrict__ bnb,
    const float* __restrict__ bnm, const float* __restrict__ bnv,
    bf16* __restrict__ gbf) {
  __shared__ bf16 sB[32 * kE];               // 16KB
  const int ptile = blockIdx.x;              // 0..1599
  const int mhalf = blockIdx.y;              // 0..1
  const int t = threadIdx.x;
  const int p0 = ptile * 32;
  {
    const int plc = t >> 3;                  // pixel 0..31
    const int co  = (t & 7) * 32;            // channel offset, 64B per thread
    async_g2l_32B(lds_off_of(&sB[plc * kE + co]),
                  &hbf[(size_t)(p0 + plc) * kE + co]);
    async_g2l_32B(lds_off_of(&sB[plc * kE + co + 16]),
                  &hbf[(size_t)(p0 + plc) * kE + co + 16]);
  }
  wait_async_all();
  __syncthreads();

  const int wave = t >> 5, lane = t & 31;
  const int mbase = mhalf * 128 + wave * 16;
  const int row   = mbase + (lane & 15);     // A row (cout)
  const int khalf = (lane >> 4) * 8;         // A K-half select
  const int kb16  = (lane >> 4) * 16;        // unused; B is indexed by khalf like A
  const int pl    = lane & 15;               // B column (pixel within subtile)

  v8f acc0 = {}, acc1 = {};
#pragma unroll
  for (int kk = 0; kk < kE; kk += 32) {
    if (kk + 32 < kE) __builtin_prefetch(&wpw[row * kE + kk + 64], 0, 0);
    v16bf a = join8(*(const v8bf*)&wpw[row * kE + kk + khalf],
                    *(const v8bf*)&wpw[row * kE + kk + khalf + 16]);
    v16bf bm0 = join8(*(const v8bf*)&sB[pl * kE + kk + khalf],
                      *(const v8bf*)&sB[pl * kE + kk + khalf + 16]);
    v16bf bm1 = join8(*(const v8bf*)&sB[(16 + pl) * kE + kk + khalf],
                      *(const v8bf*)&sB[(16 + pl) * kE + kk + khalf + 16]);
    acc0 = wmma_bf16(a, bm0, acc0);
    acc1 = wmma_bf16(a, bm1, acc1);
  }

  const int hi = lane >> 4;
  const int coutBase = mbase + hi * 8;       // 8 contiguous couts per lane
  v8bf outv0, outv1;
#pragma unroll
  for (int r = 0; r < 8; ++r) {
    const int c = coutBase + r;
    const float bias = pwb[c];
    const float inv = bng[c] * rsqrtf(bnv[c] + kBnEps);
    const float mu = bnm[c], beta = bnb[c];
    float v0 = (acc0[r] + bias - mu) * inv + beta;
    float v1 = (acc1[r] + bias - mu) * inv + beta;
    outv0[r] = (bf16)gelu_exact(v0);
    outv1[r] = (bf16)gelu_exact(v1);
  }
  *(v8bf*)&gbf[(size_t)(p0 + pl) * kE + coutBase] = outv0;
  *(v8bf*)&gbf[(size_t)(p0 + 16 + pl) * kE + coutBase] = outv1;
}

// ------------------- grouped qkv 1x1: M=768, K=32/group, scatter to q/k/v
// 16-pixel activation slice (8KB) async-staged in LDS, shared by 8 waves.
__global__ __launch_bounds__(256) void k_qkvgemm(
    const bf16* __restrict__ gbf, const bf16* __restrict__ wqkv,
    const float* __restrict__ qkvb,
    bf16* __restrict__ qb, bf16* __restrict__ kbuf, bf16* __restrict__ vb) {
  __shared__ bf16 sG[16 * kE];               // 8KB
  const int n0 = blockIdx.x * 16;
  const int t = threadIdx.x;
  {
    const int plc = t >> 4;                  // pixel 0..15
    const int co  = (t & 15) * 16;           // 32B per thread
    async_g2l_32B(lds_off_of(&sG[plc * kE + co]),
                  &gbf[(size_t)(n0 + plc) * kE + co]);
  }
  wait_async_all();
  __syncthreads();

  const int wave = t >> 5, lane = t & 31;
  const int pl = lane & 15, hi = lane >> 4;
  const int pixel = n0 + pl;
  const int b = pixel / kN;
  const int n = pixel % kN;

  for (int mt = wave; mt < 48; mt += 8) {    // 48 tiles of 16 out-channels
    const int ocBase = mt * 16;
    const int group = ocBase / 96;           // grouped-conv input group
    const int row = ocBase + pl;
    v16bf a = join8(*(const v8bf*)&wqkv[row * 32 + hi * 8],
                    *(const v8bf*)&wqkv[row * 32 + hi * 8 + 16]);
    v16bf bm = join8(*(const v8bf*)&sG[pl * kE + group * 32 + hi * 8],
                     *(const v8bf*)&sG[pl * kE + group * 32 + hi * 8 + 16]);
    v8f acc = {};
    acc = wmma_bf16(a, bm, acc);
#pragma unroll
    for (int r = 0; r < 8; ++r) {
      const int oc = ocBase + hi * 8 + r;
      const bf16 bv = (bf16)(acc[r] + qkvb[oc]);
      const int seg = oc >> 8;               // 0=q 1=k 2=v
      const int within = oc & 255;
      const int head = within >> 5;
      const int d = within & 31;
      if (seg == 0)
        qb[((size_t)(b * kNH + head) * kHD + d) * kN + n] = bv;      // (b,h,d,n)
      else if (seg == 1)
        kbuf[((size_t)(b * kNH + head) * kHD + d) * kN + n] = bv;    // (b,h,d,n)
      else
        vb[((size_t)(b * kNH + head) * kN + n) * kHD + d] = bv;      // (b,h,n,d)
    }
  }
}

// ---------------- S = softmax(scale * q k^T) over N=6400: block per (b,h)
__global__ __launch_bounds__(256) void k_attn_qk(
    const bf16* __restrict__ qb, const bf16* __restrict__ kb,
    bf16* __restrict__ pbf) {
  __shared__ float red[8 * 1024];            // per-wave 32x32 partials (32KB)
  const int bh = blockIdx.x;
  const bf16* q = qb + (size_t)bh * kHD * kN;
  const bf16* k = kb + (size_t)bh * kHD * kN;
  const int wave = threadIdx.x >> 5, lane = threadIdx.x & 31;
  const int pl = lane & 15, hi = lane >> 4;
  const int khalf = hi * 8, kb16 = hi * 16;

  v8f acc00 = {}, acc01 = {}, acc10 = {}, acc11 = {};
  for (int c = wave; c < kN / 32; c += 8) {  // 200 K-chunks of 32, 25/wave
    const int nb = c * 32;
    v16bf a0 = join8(*(const v8bf*)&q[(pl) * kN + nb + khalf],
                     *(const v8bf*)&q[(pl) * kN + nb + khalf + 16]);
    v16bf a1 = join8(*(const v8bf*)&q[(16 + pl) * kN + nb + khalf],
                     *(const v8bf*)&q[(16 + pl) * kN + nb + khalf + 16]);
    v16bf b0 = join8(*(const v8bf*)&k[(pl) * kN + nb + khalf],
                     *(const v8bf*)&k[(pl) * kN + nb + khalf + 16]);
    v16bf b1 = join8(*(const v8bf*)&k[(16 + pl) * kN + nb + khalf],
                     *(const v8bf*)&k[(16 + pl) * kN + nb + khalf + 16]);
    acc00 = wmma_bf16(a0, b0, acc00);
    acc01 = wmma_bf16(a0, b1, acc01);
    acc10 = wmma_bf16(a1, b0, acc10);
    acc11 = wmma_bf16(a1, b1, acc11);
  }
  float* my = &red[wave * 1024];
#pragma unroll
  for (int r = 0; r < 8; ++r) {
    my[(r + hi * 8) * 32 + pl]            = acc00[r];
    my[(r + hi * 8) * 32 + 16 + pl]       = acc01[r];
    my[(16 + r + hi * 8) * 32 + pl]       = acc10[r];
    my[(16 + r + hi * 8) * 32 + 16 + pl]  = acc11[r];
  }
  __syncthreads();
  constexpr float scale = 0.17677669529663688110f;   // 32^-0.5
  for (int idx = threadIdx.x; idx < 1024; idx += 256) {
    float s = 0.f;
#pragma unroll
    for (int wv = 0; wv < 8; ++wv) s += red[wv * 1024 + idx];
    red[idx] = s * scale;                   // each thread only rewrites cells it read
  }
  __syncthreads();
  if (threadIdx.x < 32) {                   // one row of S per lane
    const int rrow = threadIdx.x;
    float m = -1e30f;
    for (int c2 = 0; c2 < 32; ++c2) m = fmaxf(m, red[rrow * 32 + c2]);
    float sum = 0.f;
    for (int c2 = 0; c2 < 32; ++c2) sum += expf(red[rrow * 32 + c2] - m);
    const float inv = 1.0f / sum;
    for (int c2 = 0; c2 < 32; ++c2)
      pbf[(size_t)bh * 1024 + rrow * 32 + c2] =
          (bf16)(expf(red[rrow * 32 + c2] - m) * inv);
  }
}

// ---------------- O = P @ V; out += dyn[b,branch] * O -----------------------
__global__ __launch_bounds__(256) void k_attn_pv(
    const bf16* __restrict__ pbf, const bf16* __restrict__ vb,
    const float* __restrict__ dyn, int branch, float* __restrict__ out) {
  const int blk = blockIdx.x;                // (b, h, chunk-of-8-ntiles)
  const int chunk = blk % 50;
  const int h = (blk / 50) % kNH;
  const int b = blk / (50 * kNH);
  const int bh = b * kNH + h;
  const int wave = threadIdx.x >> 5, lane = threadIdx.x & 31;
  const int pl = lane & 15, hi = lane >> 4;
  const int n0 = (chunk * 8 + wave) * 16;    // 16-pixel tile per wave
  const float dynw = dyn[b * 3 + branch];

  const bf16* P = pbf + (size_t)bh * 1024;
  const int khalf = hi * 8;
  v16bf a0 = join8(*(const v8bf*)&P[pl * 32 + khalf],
                   *(const v8bf*)&P[pl * 32 + khalf + 16]);
  v16bf a1 = join8(*(const v8bf*)&P[(16 + pl) * 32 + khalf],
                   *(const v8bf*)&P[(16 + pl) * 32 + khalf + 16]);
  v16bf bm = join8(*(const v8bf*)&vb[((size_t)bh * kN + n0 + pl) * kHD + khalf],
                   *(const v8bf*)&vb[((size_t)bh * kN + n0 + pl) * kHD + khalf + 16]);
  v8f acc0 = {}, acc1 = {};
  acc0 = wmma_bf16(a0, bm, acc0);
  acc1 = wmma_bf16(a1, bm, acc1);

  const int n = n0 + pl;
  float* obase = out + (size_t)(b * kE + h * kHD) * kN + n;
#pragma unroll
  for (int r = 0; r < 8; ++r) {
    const int d0 = r + hi * 8;
    obase[(size_t)d0 * kN]        += dynw * acc0[r];
    obase[(size_t)(16 + d0) * kN] += dynw * acc1[r];
  }
}

// ---------------------------------------------------------------- launcher
extern "C" void kernel_launch(void* const* d_in, const int* in_sizes, int n_in,
                              void* d_out, int out_size, void* d_ws, size_t ws_size,
                              hipStream_t stream) {
  (void)in_sizes; (void)n_in; (void)out_size; (void)ws_size;
  const float* x    = (const float*)d_in[0];
  const float* pos  = (const float*)d_in[1];
  const float* dw0  = (const float*)d_in[2];
  const float* dw1  = (const float*)d_in[3];
  const float* dw2  = (const float*)d_in[4];
  const float* dwb  = (const float*)d_in[5];
  const float* pww  = (const float*)d_in[6];
  const float* pwb  = (const float*)d_in[7];
  const float* bng  = (const float*)d_in[8];
  const float* bnb  = (const float*)d_in[9];
  const float* bnm  = (const float*)d_in[10];
  const float* bnv  = (const float*)d_in[11];
  const float* qkvw = (const float*)d_in[12];
  const float* qkvb = (const float*)d_in[13];
  const float* fcw  = (const float*)d_in[14];
  const float* fcb  = (const float*)d_in[15];
  float* out = (float*)d_out;

  char* ws = (char*)d_ws;
  size_t off = 0;
  auto alloc = [&](size_t bytes) -> void* {
    void* p = ws + off;
    off += (bytes + 255) & ~(size_t)255;
    return p;
  };
  float* pooled = (float*)alloc((size_t)kB * kE * sizeof(float));
  float* dyn    = (float*)alloc((size_t)kB * 3 * sizeof(float));
  bf16*  wpw    = (bf16*)alloc((size_t)3 * kE * kE * sizeof(bf16));
  bf16*  wqk    = (bf16*)alloc((size_t)3 * 768 * 32 * sizeof(bf16));
  bf16*  hbf    = (bf16*)alloc((size_t)kNPIX * kE * sizeof(bf16));
  bf16*  gbf    = (bf16*)alloc((size_t)kNPIX * kE * sizeof(bf16));
  bf16*  qbuf   = (bf16*)alloc((size_t)kB * kNH * kHD * kN * sizeof(bf16));
  bf16*  kbuf   = (bf16*)alloc((size_t)kB * kNH * kHD * kN * sizeof(bf16));
  bf16*  vbuf   = (bf16*)alloc((size_t)kB * kNH * kHD * kN * sizeof(bf16));
  bf16*  pbuf   = (bf16*)alloc((size_t)kB * kNH * 1024 * sizeof(bf16));

  k_cvt_weights<<<1056, 256, 0, stream>>>(pww, qkvw, wpw, wqk);
  k_pool<<<kB * kE, 256, 0, stream>>>(x, pooled);
  k_dyn<<<kB, 256, 0, stream>>>(pooled, fcw, fcb, dyn);
  k_init_out<<<(kB * kE * kN / 4) / 256, 256, 0, stream>>>(x, out);

  const int dwGrid = kB * kE * kN / 256;     // 51200
  for (int i = 0; i < 3; ++i) {
    if (i == 0)
      k_dwconv<3><<<dwGrid, 256, 0, stream>>>(x, pos, dw0, dwb + 0 * kE, hbf);
    else if (i == 1)
      k_dwconv<5><<<dwGrid, 256, 0, stream>>>(x, pos, dw1, dwb + 1 * kE, hbf);
    else
      k_dwconv<7><<<dwGrid, 256, 0, stream>>>(x, pos, dw2, dwb + 2 * kE, hbf);

    k_pwgemm<<<dim3(kNPIX / 32, 2), 256, 0, stream>>>(
        hbf, wpw + (size_t)i * kE * kE, pwb + i * kE,
        bng + i * kE, bnb + i * kE, bnm + i * kE, bnv + i * kE, gbf);

    k_qkvgemm<<<kNPIX / 16, 256, 0, stream>>>(
        gbf, wqk + (size_t)i * 768 * 32, qkvb + i * 768, qbuf, kbuf, vbuf);

    k_attn_qk<<<kB * kNH, 256, 0, stream>>>(qbuf, kbuf, pbuf);

    k_attn_pv<<<kB * kNH * 50, 256, 0, stream>>>(pbuf, vbuf, dyn, i, out);
  }
}
